// MultiHeadAttentionCode_86818468922145
// MI455X (gfx1250) — matrix operand order repaired and checked by reading, not compiled, not run on hardware
//
#include <hip/hip_runtime.h>

// ---------------------------------------------------------------------------
// MHA forward on gfx1250 (MI455X): bf16 WMMA (v_wmma_f32_16x16x32_bf16) with
// fp32 accumulation, flash-style attention, software-pipelined GEMMs with
// async global->LDS staging of the weight tiles (ASYNCcnt path).
// ---------------------------------------------------------------------------

typedef __bf16 bf16;
typedef __attribute__((ext_vector_type(16))) __bf16 v16bf;
typedef __attribute__((ext_vector_type(8)))  __bf16 v8bf;
typedef __attribute__((ext_vector_type(8)))  float  v8f;

typedef int v4i __attribute__((ext_vector_type(4)));
typedef __attribute__((address_space(1))) v4i as1_v4i;   // global int4
typedef __attribute__((address_space(3))) v4i as3_v4i;   // LDS int4

#define D_MODEL 1024
#define NHEADS  16
#define DK      64
#define SEQ     2048
#define BATCH   4
#define MTOT    (BATCH * SEQ)          // 8192 rows

#if __has_builtin(__builtin_amdgcn_global_load_async_to_lds_b128)
#define HAVE_ASYNC_LDS 1
#else
#define HAVE_ASYNC_LDS 0
#endif

__device__ __forceinline__ void stage_copy16(const unsigned short* g,
                                             unsigned short* l) {
#if HAVE_ASYNC_LDS
  // per-lane 16B DMA into LDS, tracked by ASYNCcnt
  __builtin_amdgcn_global_load_async_to_lds_b128(
      (as1_v4i*)g, (as3_v4i*)l, 0, 0);
#else
  *(uint4*)l = *(const uint4*)g;       // VGPR round-trip fallback
#endif
}

__device__ __forceinline__ void stage_wait() {
#if HAVE_ASYNC_LDS
#if __has_builtin(__builtin_amdgcn_s_wait_asynccnt)
  __builtin_amdgcn_s_wait_asynccnt(0);
#else
  asm volatile("s_wait_asynccnt 0x0" ::: "memory");
#endif
#endif
}

__device__ __forceinline__ unsigned short f2bf(float f) {
  unsigned u = __builtin_bit_cast(unsigned, f);
  u += 0x7FFFu + ((u >> 16) & 1u);     // round-to-nearest-even
  return (unsigned short)(u >> 16);
}

__device__ __forceinline__ v8f wmma_bf16(v16bf a, v16bf b, v8f c) {
  return __builtin_amdgcn_wmma_f32_16x16x32_bf16(false, a, false, b,
                                                 (short)0, c, false, false);
}

// A fragment: 16x32 bf16 (MxK) from row-major [*, ld] at (row0, k0).
// Lane L (M = L%16, h = L/16): elem e -> K = (e<8 ? e : e+8) + 8h.
__device__ __forceinline__ v16bf load_a(const unsigned short* __restrict__ src,
                                        int ld, int row0, int k0, int lr, int lh) {
  const unsigned short* p = src + (size_t)(row0 + lr) * ld + k0 + 8 * lh;
  v8bf lo = *(const v8bf*)(p);
  v8bf hi = *(const v8bf*)(p + 16);
  return __builtin_shufflevector(lo, hi, 0,1,2,3,4,5,6,7,8,9,10,11,12,13,14,15);
}

// B fragment: 32x16 (KxN) with B[k][n] = src[n][k], src row-major [N, ld].
// Lane L (N = L%16, h = L/16): elem e -> K = k0 + (e<8 ? e : e+8) + 8h.
__device__ __forceinline__ v16bf load_b(const unsigned short* __restrict__ src,
                                        int ld, int n0, int k0, int lr, int lh) {
  const unsigned short* p = src + (size_t)(n0 + lr) * ld + k0 + 8 * lh;
  v8bf lo = *(const v8bf*)(p);
  v8bf hi = *(const v8bf*)(p + 16);
  return __builtin_shufflevector(lo, hi, 0,1,2,3,4,5,6,7,8,9,10,11,12,13,14,15);
}

// B fragment from the padded LDS tile (two 16B ds loads)
__device__ __forceinline__ v16bf load_b_lds(const unsigned short* bp) {
  v8bf blo = *(const v8bf*)(bp);
  v8bf bhi = *(const v8bf*)(bp + 16);
  return __builtin_shufflevector(blo, bhi,
            0,1,2,3,4,5,6,7,8,9,10,11,12,13,14,15);
}

// ---------------------------------------------------------------------------
// fp32 -> bf16 conversion (vectorized x4)
// ---------------------------------------------------------------------------
__global__ void cvt_bf16(const float* __restrict__ src,
                         unsigned short* __restrict__ dst, int n) {
  int i = (blockIdx.x * blockDim.x + threadIdx.x) * 4;
  if (i < n) {
    float4 f = *(const float4*)(src + i);
    ushort4 o;
    o.x = f2bf(f.x); o.y = f2bf(f.y); o.z = f2bf(f.z); o.w = f2bf(f.w);
    *(ushort4*)(dst + i) = o;
  }
}

// ---------------------------------------------------------------------------
// GEMM: y[m][n] = sum_k A[m][k] * W[n][k] (+ bias[n]).
// Block = 8 waves sharing one 64-wide n-chunk; per k-step the 64x32 weight
// slice is staged into LDS (double-buffered, async) and consumed by all waves;
// A fragments are register double-buffered; all four B fragments are
// preloaded before the WMMA burst so dscnt waits stagger.
// mode 0/1: bf16 out [(b*16+h)][s][d] ; mode 2: bf16 [(b*16+h)][d][s] (V^T);
// mode 3: f32 out [m][n] (final projection).
// ---------------------------------------------------------------------------
__global__ __launch_bounds__(256)
void gemm_proj(const unsigned short* __restrict__ A,   // [MTOT][1024] bf16
               const unsigned short* __restrict__ W,   // [1024][1024] bf16
               const float* __restrict__ bias,         // [1024] f32
               void* __restrict__ out, int mode) {
  __shared__ unsigned short Wt[2][64][40];   // 32 data halfs + 8 pad per row

  int tid  = threadIdx.x;
  int lane = tid & 31, lr = lane & 15, lh = lane >> 4;
  int wv   = tid >> 5;                        // wave in block (0..7)
  int n0   = (blockIdx.x & 15) * 64;
  int m0   = (blockIdx.x >> 4) * 128 + wv * 16;

  // cooperative copy assignment: 256 threads x 16B = 64 rows x 32 halfs
  int crow = tid >> 2, cch = tid & 3;
  const unsigned short* gsrc = W + (size_t)(n0 + crow) * D_MODEL + cch * 8;
  unsigned short* ldst[2] = { &Wt[0][crow][cch * 8], &Wt[1][crow][cch * 8] };

  stage_copy16(gsrc, ldst[0]);                // k-slice 0 -> buf 0
  stage_wait();
  __syncthreads();

  v16bf a_cur = load_a(A, D_MODEL, m0, 0, lr, lh);
  v8f acc[4] = {};

  for (int kk = 0; kk < 32; ++kk) {
    if (kk + 1 < 32)                          // stream next slice into other buf
      stage_copy16(gsrc + (kk + 1) * 32, ldst[(kk + 1) & 1]);
    int kn = (kk + 1 < 32) ? (kk + 1) : 0;
    v16bf a_nxt = load_a(A, D_MODEL, m0, kn * 32, lr, lh);

    // preload all four B fragments, then burst the WMMAs
    v16bf bf0 = load_b_lds(&Wt[kk & 1][ 0 + lr][8 * lh]);
    v16bf bf1 = load_b_lds(&Wt[kk & 1][16 + lr][8 * lh]);
    v16bf bf2 = load_b_lds(&Wt[kk & 1][32 + lr][8 * lh]);
    v16bf bf3 = load_b_lds(&Wt[kk & 1][48 + lr][8 * lh]);

    acc[0] = wmma_bf16(a_cur, bf0, acc[0]);
    acc[1] = wmma_bf16(a_cur, bf1, acc[1]);
    acc[2] = wmma_bf16(a_cur, bf2, acc[2]);
    acc[3] = wmma_bf16(a_cur, bf3, acc[3]);

    a_cur = a_nxt;
    stage_wait();
    __syncthreads();
  }

#pragma unroll
  for (int t = 0; t < 4; ++t) {
    int n = n0 + 16 * t + lr;
    float bv = bias[n];
#pragma unroll
    for (int r = 0; r < 8; ++r) {
      int m = m0 + 8 * lh + r;                // C layout: lane n+16*(m/8), vgpr m%8
      float v = acc[t][r] + bv;
      if (mode == 3) {
        ((float*)out)[(size_t)m * D_MODEL + n] = v;
      } else {
        int b = m >> 11, s = m & 2047;        // m = b*2048 + s
        int h = n >> 6,  d = n & 63;          // n = h*64 + d
        size_t idx = (mode == 2)
            ? (((size_t)(b * NHEADS + h) * DK + d) * SEQ + s)
            : (((size_t)(b * NHEADS + h) * SEQ + s) * DK + d);
        ((unsigned short*)out)[idx] = f2bf(v);
      }
    }
  }
}

// ---------------------------------------------------------------------------
// Flash attention: one wave per (b, h, 16-query tile).
// S^T = K·Q^T so each lane owns one query row's scores; the D-layout of two
// consecutive 16-key S^T tiles equals the 16x32 A-fragment layout of P, so P
// feeds the P·V WMMA with zero cross-lane movement. K fragments for the next
// 32-key step and V fragments for the current step are issued early so the
// softmax VALU work hides their latency.
// ---------------------------------------------------------------------------
__global__ __launch_bounds__(256)
void flash_attn(const unsigned short* __restrict__ Qb,   // [64][2048][64] bf16
                const unsigned short* __restrict__ Kb,   // [64][2048][64] bf16
                const unsigned short* __restrict__ Vtb,  // [64][64][2048] bf16
                unsigned short* __restrict__ ctxb) {     // [8192][1024] bf16
  int lane = threadIdx.x & 31;
  int lr = lane & 15, lh = lane >> 4;
  int wid = blockIdx.x * (blockDim.x >> 5) + (threadIdx.x >> 5);
  int bh = wid >> 7;                   // 0..63
  int q0 = (wid & 127) * 16;
  int b = bh >> 4, h = bh & 15;

  const unsigned short* Qh = Qb  + (size_t)bh * SEQ * DK;
  const unsigned short* Kh = Kb  + (size_t)bh * SEQ * DK;
  const unsigned short* Vh = Vtb + (size_t)bh * DK * SEQ;

  v16bf bq0 = load_b(Qh, DK, q0, 0,  lr, lh);
  v16bf bq1 = load_b(Qh, DK, q0, 32, lr, lh);

  // preload K fragments for jb = 0
  v16bf ka0 = load_a(Kh, DK, 0,  0,  lr, lh);
  v16bf ka1 = load_a(Kh, DK, 0,  32, lr, lh);
  v16bf ka2 = load_a(Kh, DK, 16, 0,  lr, lh);
  v16bf ka3 = load_a(Kh, DK, 16, 32, lr, lh);

  v8f ctx[4] = {};
  float mi = -1e30f, li = 0.f;
  const float scale = 0.125f;          // 1/sqrt(64)

  for (int jb = 0; jb < SEQ; jb += 32) {
    // V fragments for this step: issue now, consume after softmax
    v16bf bv0 = load_b(Vh, SEQ, 0,  jb, lr, lh);
    v16bf bv1 = load_b(Vh, SEQ, 16, jb, lr, lh);
    v16bf bv2 = load_b(Vh, SEQ, 32, jb, lr, lh);
    v16bf bv3 = load_b(Vh, SEQ, 48, jb, lr, lh);

    v8f s0 = {}, s1 = {};
    s0 = wmma_bf16(ka0, bq0, s0);
    s0 = wmma_bf16(ka1, bq1, s0);
    s1 = wmma_bf16(ka2, bq0, s1);
    s1 = wmma_bf16(ka3, bq1, s1);

    // prefetch K fragments for next step (clamped on last iteration)
    int jn = (jb + 32 < SEQ) ? (jb + 32) : 0;
    v16bf kn0 = load_a(Kh, DK, jn,      0,  lr, lh);
    v16bf kn1 = load_a(Kh, DK, jn,      32, lr, lh);
    v16bf kn2 = load_a(Kh, DK, jn + 16, 0,  lr, lh);
    v16bf kn3 = load_a(Kh, DK, jn + 16, 32, lr, lh);

    float mloc = -1e30f;
#pragma unroll
    for (int r = 0; r < 8; ++r) {
      s0[r] *= scale; s1[r] *= scale;
      mloc = fmaxf(mloc, fmaxf(s0[r], s1[r]));
    }
    mloc = fmaxf(mloc, __shfl_xor(mloc, 16));       // combine key halves
    float mnew  = fmaxf(mi, mloc);
    float alpha = __expf(mi - mnew);
    float p0[8], p1[8], lsum = 0.f;
#pragma unroll
    for (int r = 0; r < 8; ++r) {
      p0[r] = __expf(s0[r] - mnew);
      p1[r] = __expf(s1[r] - mnew);
      lsum += p0[r] + p1[r];
    }
    lsum += __shfl_xor(lsum, 16);
    li = li * alpha + lsum;
    mi = mnew;

    // rescale ctx accumulators (ctx lane holds queries q = r + 8*lh)
#pragma unroll
    for (int r = 0; r < 8; ++r) {
      float aq = __shfl(alpha, r + 8 * lh);
      ctx[0][r] *= aq; ctx[1][r] *= aq; ctx[2][r] *= aq; ctx[3][r] *= aq;
    }

    // P as A fragment: e<8 -> tile0 vgpr e, e>=8 -> tile1 vgpr e-8 (same lane)
    v16bf pa;
#pragma unroll
    for (int e = 0; e < 8; ++e) {
      pa[e]     = __builtin_bit_cast(bf16, f2bf(p0[e]));
      pa[e + 8] = __builtin_bit_cast(bf16, f2bf(p1[e]));
    }

    ctx[0] = wmma_bf16(pa, bv0, ctx[0]);
    ctx[1] = wmma_bf16(pa, bv1, ctx[1]);
    ctx[2] = wmma_bf16(pa, bv2, ctx[2]);
    ctx[3] = wmma_bf16(pa, bv3, ctx[3]);

    ka0 = kn0; ka1 = kn1; ka2 = kn2; ka3 = kn3;
  }

  float linv = 1.0f / li;
#pragma unroll
  for (int r = 0; r < 8; ++r) {
    float lq = __shfl(linv, r + 8 * lh);
    int q = q0 + 8 * lh + r;
    size_t base = ((size_t)(b * SEQ + q)) * D_MODEL + h * DK;
#pragma unroll
    for (int t = 0; t < 4; ++t)
      ctxb[base + t * 16 + lr] = f2bf(ctx[t][r] * lq);
  }
}

// ---------------------------------------------------------------------------
extern "C" void kernel_launch(void* const* d_in, const int* in_sizes, int n_in,
                              void* d_out, int out_size, void* d_ws, size_t ws_size,
                              hipStream_t stream) {
  const float* x  = (const float*)d_in[0];
  const float* Wq = (const float*)d_in[1];
  const float* bq = (const float*)d_in[2];
  const float* Wk = (const float*)d_in[3];
  const float* bk = (const float*)d_in[4];
  const float* Wv = (const float*)d_in[5];
  const float* bv = (const float*)d_in[6];
  const float* Wo = (const float*)d_in[7];
  const float* bo = (const float*)d_in[8];

  unsigned short* xb  = (unsigned short*)d_ws;               // 8192*1024
  unsigned short* Wqb = xb  + (size_t)MTOT * D_MODEL;        // 1024*1024 each
  unsigned short* Wkb = Wqb + (size_t)D_MODEL * D_MODEL;
  unsigned short* Wvb = Wkb + (size_t)D_MODEL * D_MODEL;
  unsigned short* Wob = Wvb + (size_t)D_MODEL * D_MODEL;
  unsigned short* Qb  = Wob + (size_t)D_MODEL * D_MODEL;     // 8192*1024 each
  unsigned short* Kb  = Qb  + (size_t)MTOT * D_MODEL;
  unsigned short* Vtb = Kb  + (size_t)MTOT * D_MODEL;
  unsigned short* Cb  = Vtb + (size_t)MTOT * D_MODEL;

  const int nx = MTOT * D_MODEL;       // 8388608
  const int nw = D_MODEL * D_MODEL;    // 1048576
  cvt_bf16<<<(nx / 4 + 255) / 256, 256, 0, stream>>>(x,  xb,  nx);
  cvt_bf16<<<(nw / 4 + 255) / 256, 256, 0, stream>>>(Wq, Wqb, nw);
  cvt_bf16<<<(nw / 4 + 255) / 256, 256, 0, stream>>>(Wk, Wkb, nw);
  cvt_bf16<<<(nw / 4 + 255) / 256, 256, 0, stream>>>(Wv, Wvb, nw);
  cvt_bf16<<<(nw / 4 + 255) / 256, 256, 0, stream>>>(Wo, Wob, nw);

  // 64 m-blocks x 16 n-chunks = 1024 blocks of 8 waves
  gemm_proj<<<1024, 256, 0, stream>>>(xb, Wqb, bq, (void*)Qb,  0);
  gemm_proj<<<1024, 256, 0, stream>>>(xb, Wkb, bk, (void*)Kb,  1);
  gemm_proj<<<1024, 256, 0, stream>>>(xb, Wvb, bv, (void*)Vtb, 2);

  // 4*16*128 = 8192 wave-tiles -> 1024 blocks
  flash_attn<<<1024, 256, 0, stream>>>(Qb, Kb, Vtb, Cb);

  gemm_proj<<<1024, 256, 0, stream>>>(Cb, Wob, bo, d_out, 3);
}
